// MNIST_GNN_19559281066611
// MI455X (gfx1250) — matrix operand order repaired and checked by reading, not compiled, not run on hardware
//
#include <hip/hip_runtime.h>
#include <hip/hip_bf16.h>

#define NN 100000
#define EE 1200000
#define GG 128
#define HH 64
#define CC 10
#define EPSV 1e-5f

typedef __attribute__((ext_vector_type(16))) _Float16 v16h;
typedef __attribute__((ext_vector_type(8)))  float    v8f;

// ---------------- degree / norm ----------------
__global__ void k_deg_init(float* deg) {
    int i = blockIdx.x * blockDim.x + threadIdx.x;
    if (i < NN) deg[i] = 1.0f;   // self-loop contributes 1
}
__global__ void k_deg_edges(const int* __restrict__ dst, float* deg) {
    int e = blockIdx.x * blockDim.x + threadIdx.x;
    if (e < EE) atomicAdd(&deg[dst[e]], 1.0f);
}
__global__ void k_deg_finish(float* deg) {
    int i = blockIdx.x * blockDim.x + threadIdx.x;
    if (i < NN) deg[i] = rsqrtf(deg[i]);   // deg >= 1
}

// ---------------- layer-0 GEMM (K=3, trivially memory bound) ----------------
__global__ void k_gemm0(const float* __restrict__ x, const float* __restrict__ W0,
                        float* __restrict__ t) {
    int idx = blockIdx.x * blockDim.x + threadIdx.x;
    if (idx >= NN * HH) return;
    int n = idx >> 6, j = idx & 63;
    t[idx] = x[n * 3 + 0] * W0[0 * HH + j]
           + x[n * 3 + 1] * W0[1 * HH + j]
           + x[n * 3 + 2] * W0[2 * HH + j];
}

// ---------------- WMMA GEMM: t[N,64] = h[N,64] @ W[64,64] ----------------
// One wave per 16-row stripe: A fragments loaded ONCE, reused across all 4
// column tiles (4x less A traffic). 8 v_wmma_f32_16x16x32_f16 per wave.
__global__ void k_gemm_wmma(const float* __restrict__ A, const float* __restrict__ W,
                            float* __restrict__ out) {
    int wave = (blockIdx.x * blockDim.x + threadIdx.x) >> 5;
    int lane = threadIdx.x & 31;
    int rowBlk = wave;                          // 6250 row blocks (exact: N = 16*6250)
    if (rowBlk >= NN / 16) return;

    int m     = rowBlk * 16 + (lane & 15);      // A row for this lane
    int nlane = lane & 15;                      // B/C column within tile
    int khalf = (lane >> 4) * 8;                // A: K sub-chunk select
    int kbb   = (lane >> 4) * 16;               // unused; B selects K with khalf, like A

    // ---- load A fragments once (K = 0..31 and K = 32..63) ----
    v16h a[2];
    #pragma unroll
    for (int h = 0; h < 2; ++h) {
        int kb = h * 32;
        #pragma unroll
        for (int j = 0; j < 8; ++j) {
            int k0 = kb + khalf + j;            // elems 0..7  : K = khalf+j
            int k1 = kb + 16 + khalf + j;       // elems 8..15 : K = 16+khalf+j
            a[h][j]     = (_Float16)A[m * HH + k0];
            a[h][j + 8] = (_Float16)A[m * HH + k1];
        }
    }

    int rbase = rowBlk * 16 + ((lane >> 4) ? 8 : 0);

    // ---- 4 column tiles, reusing A fragments ----
    #pragma unroll
    for (int colBlk = 0; colBlk < 4; ++colBlk) {
        int ncol = colBlk * 16 + nlane;
        v8f c = {};
        #pragma unroll
        for (int h = 0; h < 2; ++h) {
            int kb = h * 32;
            v16h b;
            #pragma unroll
            for (int j = 0; j < 16; ++j) {
                int k = kb + khalf + (j < 8 ? j : j + 8);   // B elem j : K = khalf+j (j<8), 16+khalf+j-8 (j>=8)
                b[j] = (_Float16)W[k * HH + ncol];
            }
            c = __builtin_amdgcn_wmma_f32_16x16x32_f16(
                    false, a[h], false, b, (short)0, c, false, false);
        }
        #pragma unroll
        for (int r = 0; r < 8; ++r)
            out[(rbase + r) * HH + ncol] = c[r];
    }
}

// ---------------- aggregation ----------------
// ag = t * dinv^2 (self-loop) + bias  — also (re)initializes scatter target
__global__ void k_agg_init(const float* __restrict__ t, const float* __restrict__ dinv,
                           const float* __restrict__ bias, float* __restrict__ ag) {
    int idx = blockIdx.x * blockDim.x + threadIdx.x;
    if (idx >= NN * HH) return;
    int n = idx >> 6, f = idx & 63;
    float dv = dinv[n];
    ag[idx] = t[idx] * dv * dv + bias[f];
}
__global__ void k_edge_scatter(const int* __restrict__ src, const int* __restrict__ dst,
                               const float* __restrict__ t, const float* __restrict__ dinv,
                               float* __restrict__ ag) {
    long long idx = (long long)blockIdx.x * blockDim.x + threadIdx.x;
    if (idx >= (long long)EE * HH) return;
    int e = (int)(idx >> 6), f = (int)(idx & 63);
    int s = src[e], d = dst[e];
    float nrm = dinv[s] * dinv[d];
    atomicAdd(&ag[(long long)d * HH + f], t[(long long)s * HH + f] * nrm);
}

// ---------------- BN (+ optional ReLU), in place ----------------
__global__ void k_bn(float* __restrict__ h, const float* __restrict__ gamma,
                     const float* __restrict__ beta, const float* __restrict__ mean,
                     const float* __restrict__ var, int do_relu) {
    int idx = blockIdx.x * blockDim.x + threadIdx.x;
    if (idx >= NN * HH) return;
    int f = idx & 63;
    float v = (h[idx] - mean[f]) * (gamma[f] * rsqrtf(var[f] + EPSV)) + beta[f];
    if (do_relu) v = fmaxf(v, 0.0f);
    h[idx] = v;
}

// ---------------- pooling ----------------
__global__ void k_pool_init(float* psum, float* pmax, float* pcnt) {
    int i = blockIdx.x * blockDim.x + threadIdx.x;
    if (i < GG * HH) { psum[i] = 0.0f; pmax[i] = -INFINITY; }
    if (i < GG) pcnt[i] = 0.0f;
}
__device__ inline void atomicMaxFloat(float* addr, float v) {
    if (v >= 0.0f) atomicMax((int*)addr, __float_as_int(v));
    else           atomicMin((unsigned int*)addr, __float_as_uint(v));
}
__global__ void k_pool_scatter(const float* __restrict__ h, const int* __restrict__ batch,
                               float* psum, float* pmax, float* pcnt) {
    int idx = blockIdx.x * blockDim.x + threadIdx.x;
    if (idx >= NN * HH) return;
    int n = idx >> 6, f = idx & 63;
    int g = batch[n];
    float v = h[idx];
    atomicAdd(&psum[g * HH + f], v);
    atomicMaxFloat(&pmax[g * HH + f], v);
    if (f == 0) atomicAdd(&pcnt[g], 1.0f);
}

// ---------------- classifier head (tiny) ----------------
__global__ void k_classifier(const float* __restrict__ psum, const float* __restrict__ pmax,
                             const float* __restrict__ pcnt,
                             const float* __restrict__ Wc1, const float* __restrict__ bc1,
                             const float* __restrict__ Wc2, const float* __restrict__ bc2,
                             float* __restrict__ out) {
    __shared__ float gv[2 * HH];
    __shared__ float o1[HH];
    __shared__ float logits[CC];
    int g = blockIdx.x, t = threadIdx.x;      // 64 threads
    float invc = 1.0f / pcnt[g];
    gv[t]       = psum[g * HH + t] * invc;    // mean
    gv[HH + t]  = pmax[g * HH + t];           // max
    __syncthreads();
    float acc = bc1[t];
    #pragma unroll 4
    for (int k = 0; k < 2 * HH; ++k) acc += gv[k] * Wc1[k * HH + t];
    o1[t] = fmaxf(acc, 0.0f);
    __syncthreads();
    if (t < CC) {
        float acc2 = bc2[t];
        for (int j = 0; j < HH; ++j) acc2 += o1[j] * Wc2[j * CC + t];
        logits[t] = acc2;
    }
    __syncthreads();
    if (t == 0) {
        float mx = logits[0];
        for (int c = 1; c < CC; ++c) mx = fmaxf(mx, logits[c]);
        float se = 0.0f;
        for (int c = 0; c < CC; ++c) se += expf(logits[c] - mx);
        float lse = mx + logf(se);
        for (int c = 0; c < CC; ++c) out[g * CC + c] = logits[c] - lse;
    }
}

extern "C" void kernel_launch(void* const* d_in, const int* in_sizes, int n_in,
                              void* d_out, int out_size, void* d_ws, size_t ws_size,
                              hipStream_t stream) {
    const float* x     = (const float*)d_in[0];
    const int*   eidx  = (const int*)d_in[1];
    const int*   batch = (const int*)d_in[2];
    // d_in[3] = num_graphs (unused; GG hardcoded)
    const float* W0    = (const float*)d_in[4];
    const float* Ws    = (const float*)d_in[5];   // [2,64,64]
    const float* bs    = (const float*)d_in[6];   // [3,64]
    const float* gamma = (const float*)d_in[7];
    const float* beta  = (const float*)d_in[8];
    const float* rmean = (const float*)d_in[9];
    const float* rvar  = (const float*)d_in[10];
    const float* Wc1   = (const float*)d_in[11];
    const float* bc1   = (const float*)d_in[12];
    const float* Wc2   = (const float*)d_in[13];
    const float* bc2   = (const float*)d_in[14];
    float* out = (float*)d_out;

    const int* src = eidx;
    const int* dst = eidx + EE;

    float* ws   = (float*)d_ws;
    float* dinv = ws;
    float* tb   = dinv + NN;
    float* ag   = tb + (size_t)NN * HH;
    float* psum = ag + (size_t)NN * HH;
    float* pmax = psum + GG * HH;
    float* pcnt = pmax + GG * HH;

    const int B = 256;
    const int gN   = (NN + B - 1) / B;
    const int gE   = (EE + B - 1) / B;
    const int gNH  = (NN * HH + B - 1) / B;          // 25000
    const long long EH = (long long)EE * HH;
    const int gEH  = (int)((EH + B - 1) / B);        // 300000
    const int wavesW = NN / 16;                      // 6250 waves, 1 per row stripe
    const int gWMMA  = (wavesW + (B / 32) - 1) / (B / 32);   // 782 blocks

    // normalization
    k_deg_init  <<<gN, B, 0, stream>>>(dinv);
    k_deg_edges <<<gE, B, 0, stream>>>(dst, dinv);
    k_deg_finish<<<gN, B, 0, stream>>>(dinv);

    for (int l = 0; l < 3; ++l) {
        if (l == 0)
            k_gemm0<<<gNH, B, 0, stream>>>(x, W0, tb);
        else
            k_gemm_wmma<<<gWMMA, B, 0, stream>>>(ag, Ws + (size_t)(l - 1) * HH * HH, tb);
        k_agg_init    <<<gNH, B, 0, stream>>>(tb, dinv, bs + l * HH, ag);
        k_edge_scatter<<<gEH, B, 0, stream>>>(src, dst, tb, dinv, ag);
        k_bn          <<<gNH, B, 0, stream>>>(ag, gamma + l * HH, beta + l * HH,
                                              rmean + l * HH, rvar + l * HH, (l < 2) ? 1 : 0);
    }

    k_pool_init   <<<(GG * HH + B - 1) / B, B, 0, stream>>>(psum, pmax, pcnt);
    k_pool_scatter<<<gNH, B, 0, stream>>>(ag, batch, psum, pmax, pcnt);
    k_classifier  <<<GG, HH, 0, stream>>>(psum, pmax, pcnt, Wc1, bc1, Wc2, bc2, out);
}
